// TemporalContrastiveSAE_16569983828629
// MI455X (gfx1250) — matrix operand order repaired and checked by reading, not compiled, not run on hardware
//
#include <hip/hip_runtime.h>
#include <hip/hip_bf16.h>

#define D_IN   4096
#define D_SAE  32768
#define B_SZ   4096
#define K_TOP  64

typedef __attribute__((ext_vector_type(16))) __bf16 v16bf;
typedef __attribute__((ext_vector_type(8)))  __bf16 v8bf;
typedef __attribute__((ext_vector_type(8)))  float  v8f;

// ---------------- Encode GEMM: pre = ReLU((x - b_dec) @ W_enc + b_enc) ----
// bf16 WMMA (v_wmma_f32_16x16x32_bf16), f32 accumulate.
// Tile: BM=128 x BN=128, BK=32. 256 threads = 8 waves arranged 2(M) x 4(N),
// each wave owns a 64x32 sub-tile = 4x2 wmma accumulators.
// Single-stage software pipeline; all 6 fragments hoisted into registers so
// the 8 WMMAs issue without per-pair ds waits.

#define BM 128
#define BN 128
#define BK 32
#define LDT 40   // padded LDS row stride (bf16 elems); 80B keeps 16B alignment

__device__ inline v16bf cat8(v8bf lo, v8bf hi) {
    v16bf r;
#pragma unroll
    for (int i = 0; i < 8; ++i) { r[i] = lo[i]; r[i + 8] = hi[i]; }
    return r;
}

__device__ inline unsigned pack_bf16(float a, float b) {
    union { __bf16 h[2]; unsigned u; } p;
    p.h[0] = (__bf16)a; p.h[1] = (__bf16)b;
    return p.u;
}

__global__ __launch_bounds__(256)
void sae_encode_gemm(const float* __restrict__ x,
                     const float* __restrict__ W_enc,
                     const float* __restrict__ b_enc,
                     const float* __restrict__ b_dec,
                     float* __restrict__ pre)   // (B, D_SAE)
{
    __shared__ __bf16 As[BM * LDT];  // [m][k], k-contiguous
    __shared__ __bf16 Ws[BN * LDT];  // [n][k], k-contiguous (transposed stage)

    const int tid   = threadIdx.x;
    const int wave  = tid >> 5;
    const int lane  = tid & 31;
    const int waveM = wave >> 2;     // 0..1
    const int waveN = wave & 3;      // 0..3
    const int l15   = lane & 15;
    const int blockN = blockIdx.x * BN;
    const int blockM = blockIdx.y * BM;

    v8f acc[4][2];
#pragma unroll
    for (int mt = 0; mt < 4; ++mt)
#pragma unroll
        for (int nt = 0; nt < 2; ++nt)
#pragma unroll
            for (int r = 0; r < 8; ++r) acc[mt][nt][r] = 0.0f;

    const int kbA = (lane < 16) ? 0 : 8;    // A frag K base
    const int kbB = (lane < 16) ? 0 : 8;    // B frag K base

    // ---- pipeline registers ----
    float4 ax[4], ab[4];
    float4 w0[2], w1[2];

    // per-thread static addressing for the staging work
    const int amq[4]  = { (tid * 4 + 0) >> 3, (tid * 4 + 1) >> 3,
                          (tid * 4 + 2) >> 3, (tid * 4 + 3) >> 3 };
    const int akq[4]  = { ((tid * 4 + 0) & 7) * 4, ((tid * 4 + 1) & 7) * 4,
                          ((tid * 4 + 2) & 7) * 4, ((tid * 4 + 3) & 7) * 4 };
    const int wkp[2]  = { (tid * 2 + 0) >> 5, (tid * 2 + 1) >> 5 };      // k-pair 0..15
    const int wnq[2]  = { ((tid * 2 + 0) & 31) * 4, ((tid * 2 + 1) & 31) * 4 };

    auto load_tiles = [&](int k0) {
#pragma unroll
        for (int i = 0; i < 4; ++i) {
            ax[i] = *(const float4*)(x + (size_t)(blockM + amq[i]) * D_IN + k0 + akq[i]);
            ab[i] = *(const float4*)(b_dec + k0 + akq[i]);
        }
#pragma unroll
        for (int j = 0; j < 2; ++j) {
            const float* base = W_enc + (size_t)(k0 + 2 * wkp[j]) * D_SAE + blockN + wnq[j];
            w0[j] = *(const float4*)base;
            w1[j] = *(const float4*)(base + D_SAE);
        }
    };

    auto store_tiles = [&]() {
#pragma unroll
        for (int i = 0; i < 4; ++i) {
            uint2 p;
            p.x = pack_bf16(ax[i].x - ab[i].x, ax[i].y - ab[i].y);
            p.y = pack_bf16(ax[i].z - ab[i].z, ax[i].w - ab[i].w);
            *(uint2*)&As[amq[i] * LDT + akq[i]] = p;   // 8B packed store
        }
#pragma unroll
        for (int j = 0; j < 2; ++j) {
            const float* f0 = (const float*)&w0[j];
            const float* f1 = (const float*)&w1[j];
#pragma unroll
            for (int n = 0; n < 4; ++n) {
                // (k, k+1) pair for column nq+n -> one 32-bit LDS store
                *(unsigned*)&Ws[(wnq[j] + n) * LDT + 2 * wkp[j]] = pack_bf16(f0[n], f1[n]);
            }
        }
    };

    load_tiles(0);

    for (int k0 = 0; k0 < D_IN; k0 += BK) {
        store_tiles();
        __syncthreads();

        // issue next tile's global loads before the matrix math (pipeline)
        if (k0 + BK < D_IN) {
            load_tiles(k0 + BK);
            // prefetch W panel two K-steps ahead into L2 (global_prefetch_b8)
            if (k0 + 2 * BK < D_IN)
                __builtin_prefetch(W_enc + (size_t)(k0 + 2 * BK + 2 * wkp[0]) * D_SAE
                                        + blockN + wnq[0], 0, 0);
        }

        // ---- hoist ALL fragments into distinct registers first ----
        v16bf bfrag[2];
#pragma unroll
        for (int nt = 0; nt < 2; ++nt) {
            const __bf16* bp = &Ws[(waveN * 32 + nt * 16 + l15) * LDT + kbB];
            bfrag[nt] = cat8(*(const v8bf*)bp, *(const v8bf*)(bp + 16));
        }
        v16bf afrag[4];
#pragma unroll
        for (int mt = 0; mt < 4; ++mt) {
            const __bf16* ap = &As[(waveM * 64 + mt * 16 + l15) * LDT + kbA];
            afrag[mt] = cat8(*(const v8bf*)ap, *(const v8bf*)(ap + 16));
        }
        // ---- 8 back-to-back WMMAs ----
#pragma unroll
        for (int mt = 0; mt < 4; ++mt)
#pragma unroll
            for (int nt = 0; nt < 2; ++nt) {
                acc[mt][nt] = __builtin_amdgcn_wmma_f32_16x16x32_bf16(
                    false, afrag[mt], false, bfrag[nt], (short)0, acc[mt][nt], false, false);
            }
        __syncthreads();
    }

    // ---- Epilogue: + b_enc, ReLU, store pre-activations ----
    const int rowoff = (lane >= 16) ? 8 : 0;
#pragma unroll
    for (int nt = 0; nt < 2; ++nt) {
        const int col = blockN + waveN * 32 + nt * 16 + l15;
        const float be = b_enc[col];
#pragma unroll
        for (int mt = 0; mt < 4; ++mt) {
            const int mbase = blockM + waveM * 64 + mt * 16 + rowoff;
#pragma unroll
            for (int r = 0; r < 8; ++r) {
                float v = acc[mt][nt][r] + be;
                pre[(size_t)(mbase + r) * D_SAE + col] = v > 0.0f ? v : 0.0f;
            }
        }
    }
}

// ---------------- Top-K per row (exact 3-level radix select) --------------
// pre-activations are >= 0, so float order == uint-bit order.
// In-place: reads dense pre row from z region, writes back sparse z row.
__global__ __launch_bounds__(256)
void sae_topk(float* __restrict__ z,      // (B, D_SAE) in/out
              float* __restrict__ vals,   // (B, K)
              int*   __restrict__ idx)    // (B, K)
{
    __shared__ unsigned rowb[D_SAE];          // 128 KB: full row in LDS
    __shared__ unsigned hist[2048];           // 8 KB
    __shared__ unsigned bitmap[D_SAE / 32];   // 4 KB
    __shared__ unsigned sT0, sT1, sT2;
    __shared__ int      sRemain;
    __shared__ unsigned cnt_sel, cnt_tie;

    const int tid = threadIdx.x;
    const int row = blockIdx.x;
    float* zrow = z + (size_t)row * D_SAE;

    for (int i = tid; i < D_SAE / 32; i += 256) bitmap[i] = 0;
    if (tid == 0) { cnt_sel = 0; cnt_tie = 0; }
    for (int i = tid; i < D_SAE; i += 256) rowb[i] = __float_as_uint(zrow[i]);

    // Level 0: bits [30:20] (11 bits; sign bit is 0)
    for (int i = tid; i < 2048; i += 256) hist[i] = 0;
    __syncthreads();
    for (int i = tid; i < D_SAE; i += 256) atomicAdd(&hist[rowb[i] >> 20], 1u);
    __syncthreads();
    if (tid == 0) {
        unsigned above = 0; int b;
        for (b = 2047; b >= 0; --b) {
            if (above + hist[b] >= (unsigned)K_TOP) break;
            above += hist[b];
        }
        sT0 = (unsigned)b; sRemain = K_TOP - (int)above;
    }
    __syncthreads();
    const unsigned T0 = sT0;
    // Level 1: bits [19:9] within bucket T0
    for (int i = tid; i < 2048; i += 256) hist[i] = 0;
    __syncthreads();
    for (int i = tid; i < D_SAE; i += 256) {
        const unsigned u = rowb[i];
        if ((u >> 20) == T0) atomicAdd(&hist[(u >> 9) & 0x7FF], 1u);
    }
    __syncthreads();
    if (tid == 0) {
        const int need = sRemain;
        unsigned above = 0; int b;
        for (b = 2047; b >= 0; --b) {
            if (above + hist[b] >= (unsigned)need) break;
            above += hist[b];
        }
        sT1 = (unsigned)b; sRemain = need - (int)above;
    }
    __syncthreads();
    const unsigned pref = (T0 << 11) | sT1;
    // Level 2: bits [8:0] within (T0,T1)
    for (int i = tid; i < 512; i += 256) hist[i] = 0;
    __syncthreads();
    for (int i = tid; i < D_SAE; i += 256) {
        const unsigned u = rowb[i];
        if ((u >> 9) == pref) atomicAdd(&hist[u & 0x1FF], 1u);
    }
    __syncthreads();
    if (tid == 0) {
        const int need = sRemain;
        unsigned above = 0; int b;
        for (b = 511; b >= 0; --b) {
            if (above + hist[b] >= (unsigned)need) break;
            above += hist[b];
        }
        sT2 = (unsigned)b; sRemain = need - (int)above;
    }
    __syncthreads();
    const unsigned thrbits = ((T0 << 11 | sT1) << 9) | sT2; // exact K-th value bits
    const int tiesNeeded = sRemain;

    // Selection: strictly greater -> in; equal -> first `tiesNeeded` claimers.
    for (int i = tid; i < D_SAE; i += 256) {
        const unsigned u = rowb[i];
        bool sel = false;
        if (u > thrbits) sel = true;
        else if (u == thrbits) {
            const unsigned p = atomicAdd(&cnt_tie, 1u);
            if ((int)p < tiesNeeded) sel = true;
        }
        if (sel) {
            const unsigned slot = atomicAdd(&cnt_sel, 1u);   // 0..K-1
            vals[(size_t)row * K_TOP + slot] = __uint_as_float(u);
            idx [(size_t)row * K_TOP + slot] = i;
            atomicOr(&bitmap[i >> 5], 1u << (i & 31));
        }
    }
    __syncthreads();

    // Write sparse z row (dense output, mostly zeros)
    for (int i = tid; i < D_SAE; i += 256) {
        const bool on = (bitmap[i >> 5] >> (i & 31)) & 1u;
        zrow[i] = on ? __uint_as_float(rowb[i]) : 0.0f;
    }
}

// ---------------- Sparse decode: x_hat = sum_k val[k]*W_dec[idx[k],:] + b_dec
__global__ __launch_bounds__(256)
void sae_decode(const float* __restrict__ vals,
                const int*   __restrict__ idx,
                const float* __restrict__ W_dec,
                const float* __restrict__ b_dec,
                float* __restrict__ xhat)
{
    __shared__ float sv[K_TOP];
    __shared__ int   si[K_TOP];
    const int tid = threadIdx.x;
    const int row = blockIdx.x;
    if (tid < K_TOP) {
        sv[tid] = vals[(size_t)row * K_TOP + tid];
        si[tid] = idx [(size_t)row * K_TOP + tid];
    }
    __syncthreads();

    float acc[16];
#pragma unroll
    for (int j = 0; j < 16; ++j) acc[j] = b_dec[tid + j * 256];

    for (int k = 0; k < K_TOP; ++k) {
        const float v = sv[k];
        const float* wr = W_dec + (size_t)si[k] * D_IN;
#pragma unroll
        for (int j = 0; j < 16; ++j) acc[j] = fmaf(v, wr[tid + j * 256], acc[j]);
    }
    float* out = xhat + (size_t)row * D_IN;
#pragma unroll
    for (int j = 0; j < 16; ++j) out[tid + j * 256] = acc[j];
}

// ---------------- launch ----------------
extern "C" void kernel_launch(void* const* d_in, const int* in_sizes, int n_in,
                              void* d_out, int out_size, void* d_ws, size_t ws_size,
                              hipStream_t stream) {
    const float* x     = (const float*)d_in[0];
    const float* W_enc = (const float*)d_in[1];
    const float* b_enc = (const float*)d_in[2];
    const float* W_dec = (const float*)d_in[3];
    const float* b_dec = (const float*)d_in[4];

    float* xhat = (float*)d_out;                       // (B, D_IN)
    float* z    = xhat + (size_t)B_SZ * D_IN;          // (B, D_SAE): pre -> sparse z in place

    float* vals = (float*)d_ws;                        // (B, K)
    int*   idxs = (int*)((char*)d_ws + (size_t)B_SZ * K_TOP * sizeof(float));

    dim3 g1(D_SAE / BN, B_SZ / BM);                    // 256 x 32 workgroups
    sae_encode_gemm<<<g1, 256, 0, stream>>>(x, W_enc, b_enc, b_dec, z);
    sae_topk<<<B_SZ, 256, 0, stream>>>(z, vals, idxs);
    sae_decode<<<B_SZ, 256, 0, stream>>>(vals, idxs, W_dec, b_dec, xhat);
}
